// ConsistencyMaskFromBoxes_76836964925811
// MI455X (gfx1250) — 8-bit matrix operand order repaired and checked by reading, compiled, not run on hardware
//
#include <hip/hip_runtime.h>
#include <hip/hip_bf16.h>

typedef __attribute__((ext_vector_type(8))) int v8i;

#define BN  16
#define HH  640
#define WWW 640
#define MB  256

__global__ __launch_bounds__(256) void cmfb_init(
    const float* __restrict__ bboxes, const int* __restrict__ bidx,
    const unsigned char* __restrict__ is_seg,
    float* __restrict__ accum, unsigned* __restrict__ effmask,
    uint2* __restrict__ boxpack)
{
    __shared__ unsigned smask;
    const int t = threadIdx.x;
    if (t == 0) smask = 0u;
    __syncthreads();
    {
        const float cx = bboxes[4*t + 0] * (float)WWW;
        const float cy = bboxes[4*t + 1] * (float)HH;
        const float bw = bboxes[4*t + 2] * (float)WWW;
        const float bh = bboxes[4*t + 3] * (float)HH;
        const int x1 = (int)fminf(fmaxf(cx - bw * 0.5f, 0.f), (float)(WWW - 1));
        const int y1 = (int)fminf(fmaxf(cy - bh * 0.5f, 0.f), (float)(HH  - 1));
        const int x2 = (int)fminf(fmaxf(cx + bw * 0.5f, 0.f), (float)(WWW - 1));
        const int y2 = (int)fminf(fmaxf(cy + bh * 0.5f, 0.f), (float)(HH  - 1));
        const int idx = bidx[t];
        const unsigned xp = (unsigned)x1 | ((unsigned)x2 << 16);
        const unsigned yp = (unsigned)y1 | ((unsigned)y2 << 10) | ((unsigned)(idx & 15) << 20);
        boxpack[t] = make_uint2(xp, yp);
        if (idx >= 0 && idx < BN) atomicOr(&smask, 1u << idx);
    }
    __syncthreads();
    if (t == 0) {
        const unsigned hb = smask;
        unsigned em = 0u;
        for (int i = 0; i < BN; ++i)
            if (!is_seg[i] && ((hb >> i) & 1u)) em |= (1u << i);
        *effmask = em;
        *accum = 0.f;
    }
}

__global__ __launch_bounds__(256) void cmfb_main(
    const float* __restrict__ seg, const uint2* __restrict__ boxpack,
    const unsigned* __restrict__ effmask, float* __restrict__ accum)
{
    __shared__ uint2 sbox[MB];
    __shared__ __align__(16) int sA[1024];
    __shared__ __align__(16) int sB[8192];
    const int t      = threadIdx.x;
    const int bimg   = blockIdx.z;
    const int ybase  = blockIdx.y * 16;
    const int xbase0 = blockIdx.x * 128;

    sbox[t] = boxpack[t];
    __syncthreads();

    for (int d = t; d < 1024; d += 256) {
        const int kt = d >> 8, lane = (d >> 3) & 31, v = d & 7;
        const int y  = ybase + (lane & 15);
        const int kb = kt * 64 + ((v >> 1) << 4) + ((v & 1) << 2) + ((lane >> 4) << 3);
        unsigned acc = 0u;
#pragma unroll
        for (int j = 0; j < 4; ++j) {
            const uint2 bx = sbox[kb + j];
            const int y1 = (int)(bx.y & 1023u);
            const int y2 = (int)((bx.y >> 10) & 1023u);
            const int ix = (int)((bx.y >> 20) & 15u);
            const unsigned cnd = (ix == bimg && y >= y1 && y <= y2) ? 1u : 0u;
            acc |= cnd << (8 * j);
        }
        sA[d] = (int)acc;
    }
    for (int d = t; d < 8192; d += 256) {
        const int xt = d >> 10, rem = d & 1023;
        const int kt = rem >> 8, lane = (rem >> 3) & 31, v = rem & 7;
        const int x  = xbase0 + xt * 16 + (lane & 15);
        const int kb = kt * 64 + ((v >> 1) << 4) + ((v & 1) << 2) + ((lane >> 4) << 3);
        unsigned acc = 0u;
#pragma unroll
        for (int j = 0; j < 4; ++j) {
            const uint2 bx = sbox[kb + j];
            const int x1 = (int)(bx.x & 0xffffu);
            const int x2 = (int)(bx.x >> 16);
            const unsigned cnd = (x >= x1 && x <= x2) ? 1u : 0u;
            acc |= cnd << (8 * j);
        }
        sB[d] = (int)acc;
    }
    __syncthreads();

    const int wave = t >> 5, lane = t & 31;
    const int x0 = xbase0 + wave * 16;
    v8i c = {};
    const int4* Ap = (const int4*)sA;
    const int4* Bp = (const int4*)sB + wave * 256;
#pragma unroll
    for (int kt = 0; kt < 4; ++kt) {
        const int4 a0 = Ap[kt * 64 + lane * 2 + 0];
        const int4 a1 = Ap[kt * 64 + lane * 2 + 1];
        const int4 b0 = Bp[kt * 64 + lane * 2 + 0];
        const int4 b1 = Bp[kt * 64 + lane * 2 + 1];
        v8i a, bv;
        a[0]=a0.x;  a[1]=a0.y;  a[2]=a0.z;  a[3]=a0.w;
        a[4]=a1.x;  a[5]=a1.y;  a[6]=a1.z;  a[7]=a1.w;
        bv[0]=b0.x; bv[1]=b0.y; bv[2]=b0.z; bv[3]=b0.w;
        bv[4]=b1.x; bv[5]=b1.y; bv[6]=b1.z; bv[7]=b1.w;
        c = __builtin_amdgcn_wmma_i32_16x16x64_iu8(false, a, false, bv, c, false, false);
    }

    const float eff = ((*effmask >> bimg) & 1u) ? 1.f : 0.f;
    const int   hi  = lane >> 4;
    const float* srow = seg + ((size_t)bimg * HH + ybase) * WWW + (x0 + (lane & 15));
    float sum = 0.f;
#pragma unroll
    for (int v = 0; v < 8; ++v) {
        const int   y  = v + hi * 8;
        const float l  = srow[(size_t)y * WWW] * eff;
        const float tf = (c[v] > 0) ? eff : 0.f;
        const float bce = fmaxf(l, 0.f) - l * tf + log1pf(__expf(-fabsf(l)));
        sum += bce;
    }
    for (int off = 16; off > 0; off >>= 1) sum += __shfl_xor(sum, off, 32);
    if (lane == 0) atomicAdd(accum, sum);
}

__global__ void cmfb_fin(const float* __restrict__ accum,
                         const unsigned* __restrict__ effmask,
                         float* __restrict__ out)
{
    out[0] = (*effmask != 0u)
           ? (0.1f * (*accum) * (1.0f / ((float)BN * (float)HH * (float)WWW)))
           : 0.f;
}

extern "C" void kernel_launch(void* const* d_in, const int* in_sizes, int n_in,
                              void* d_out, int out_size, void* d_ws, size_t ws_size,
                              hipStream_t stream)
{
    const float*         seg    = (const float*)d_in[0];
    const float*         bboxes = (const float*)d_in[1];
    const int*           bidx   = (const int*)d_in[2];
    const unsigned char* is_seg = (const unsigned char*)d_in[3];

    float*    accum   = (float*)d_ws;
    unsigned* effmask = (unsigned*)((char*)d_ws + 4);
    uint2*    boxpack = (uint2*)((char*)d_ws + 16);
    float*    out     = (float*)d_out;

    cmfb_init<<<1, 256, 0, stream>>>(bboxes, bidx, is_seg, accum, effmask, boxpack);
    cmfb_main<<<dim3(5, 40, 16), 256, 0, stream>>>(seg, boxpack, effmask, accum);
    cmfb_fin<<<1, 1, 0, stream>>>(accum, effmask, out);
}
